// SSEN_6734508720421
// MI455X (gfx1250) — matrix operand order repaired and checked by reading, not compiled, not run on hardware
//
#include <hip/hip_runtime.h>

typedef _Float16 half_t;
typedef __attribute__((ext_vector_type(16))) _Float16 v16h;
typedef __attribute__((ext_vector_type(8)))  _Float16 v8h;
typedef __attribute__((ext_vector_type(8)))  float    v8f;

#define BB   16
#define HH   128
#define WWID 128
#define HWXY (HH * WWID)   // 16384 = 2^14
#define LBROW 130          // padded row: 1 + 128 + 1 halo columns

// ---------------------------------------------------------------------------
// fp32 -> fp16 cast (activations)
// ---------------------------------------------------------------------------
__global__ __launch_bounds__(256) void cast_f32_to_f16(const float* __restrict__ src,
                                                       half_t* __restrict__ dst, int n) {
    int i = blockIdx.x * 256 + threadIdx.x;
    if (i < n) dst[i] = (half_t)src[i];
}

// ---------------------------------------------------------------------------
// fp32 -> fp16 cast + K-reorder for weights:
//   src OIHW  [cout][cin][tap]  ->  dst [cout][tap][cin]   (tap-major K)
// ---------------------------------------------------------------------------
__global__ __launch_bounds__(256) void cast_permute_w(const float* __restrict__ src,
                                                      half_t* __restrict__ dst,
                                                      int Cin, int n) {
    int idx = blockIdx.x * 256 + threadIdx.x;
    if (idx >= n) return;
    const int K   = Cin * 9;
    const int cout = idx / K;
    const int rem  = idx - cout * K;
    const int tap  = rem / Cin;
    const int cin  = rem - tap * Cin;
    dst[idx] = (half_t)src[((size_t)cout * Cin + cin) * 9 + tap];
}

// ---------------------------------------------------------------------------
// 3x3 SAME conv as implicit GEMM on WMMA f16 (f32 accumulate).
//   M = Cout (64-wide tiles), N = 256 px (two image rows), K = 9*Cin tap-major.
//   Per 32-channel chunk: raw 4-row x 130-col x 32-ch halo block staged ONCE,
//   then 9 fully-unrolled taps read shifted views (immediate ds offsets) and
//   issue 12 ds_load_b128 -> 8 v_wmma each. 2 barriers per chunk total.
// ---------------------------------------------------------------------------
__global__ __launch_bounds__(256) void conv3x3_wmma(
    const half_t* __restrict__ Wt,    // f16 weights [Cout][9*Cin] (tap-major K)
    const half_t* __restrict__ in0,   // f16 [B][split][H][W]
    const half_t* __restrict__ in1,   // f16 [B][Cin-split][H][W]
    const float*  __restrict__ bias,  // [Cout] or null
    half_t* __restrict__ out16,       // f16 [B][Cout][H][W] or null
    float*  __restrict__ out32,       // f32 [B][Cout][H][W] or null
    int Cin, int split, int Cout, int leaky)
{
    __shared__ __align__(16) half_t lA[9 * 64 * 32];      // [tap][m][k]   36 KB
    __shared__ __align__(16) half_t lB[4 * LBROW * 32];   // [row][xpad][k] 32.5 KB

    const int tid = threadIdx.x;
    const int mblocks = Cout >> 6;
    int bid = blockIdx.x;
    const int nb = bid & 63;  bid >>= 6;            // 64 two-row tiles per image
    const int mb = bid % mblocks;
    const int b  = bid / mblocks;
    const int y0 = nb << 1;                         // first of the two output rows
    const int m0 = mb << 6;
    const int Ktot = Cin * 9;
    const int c1   = Cin - split;

    const int lane = tid & 31;
    const int wid  = tid >> 5;
    const int lrow = lane & 15;

    v8f acc[2][4] = {};

    // staging coordinates
    const int am = tid >> 2;            // weight row 0..63
    const int ak = (tid & 3) << 3;      // k sub-offset 0,8,16,24
    const int xst   = tid & 127;        // x column for B staging
    const int rbase = tid >> 7;         // 0 or 1

    // per-lane fragment base pointers (tap offsets become immediates)
    const int kba = (lane < 16) ? 0 : 16;    // A contiguous K-halves
    const int kbb = (lane < 16) ? 0 : 16;    // B contiguous K-halves
    const half_t* pA = &lA[lrow * 32 + kba];
    const int xc0 = (wid << 5) + lrow;
    const int xc1 = xc0 + 16;
    // base at dy=-1, dx=-1 so all tap offsets are non-negative immediates
    const half_t* pB0 = &lB[(((xc0 >> 7)) * LBROW + (xc0 & 127)) * 32 + kbb];
    const half_t* pB1 = &lB[(((xc1 >> 7)) * LBROW + (xc1 & 127)) * 32 + kbb];

    // zero the halo columns once (visible after first in-loop barrier)
    if (tid < 32) {
        const int r = tid >> 3, side = (tid >> 2) & 1, v = tid & 3;
        v8h z = {};
        *(v8h*)&lB[(r * LBROW + side * (LBROW - 1)) * 32 + v * 8] = z;
    }

    for (int c0 = 0; c0 < Cin; c0 += 32) {
        // ---- stage A: all 9 taps, 16B vector copies (16B aligned)
        {
            const half_t* gp = Wt + (size_t)(m0 + am) * Ktot + (size_t)(c0 + ak);
            #pragma unroll
            for (int tap = 0; tap < 9; ++tap)
                *(float4*)&lA[tap * 2048 + am * 32 + ak] = *(const float4*)(gp + tap * Cin);
            if (c0 + 32 < Cin)
                __builtin_prefetch(gp + 32, 0, 1);   // global_prefetch next chunk
        }
        // ---- stage B: raw 4 rows x 128 px x 32 ch (halo rows zeroed)
        {
            const half_t* src0 = (c0 < split)
                ? in0 + ((size_t)b * split + c0) * HWXY
                : in1 + ((size_t)b * c1 + (c0 - split)) * HWXY;
            #pragma unroll
            for (int j = 0; j < 2; ++j) {
                const int r  = rbase + 2 * j;        // 0..3
                const int ys = y0 - 1 + r;
                v8h t[4] = {};
                if (ys >= 0 && ys < HH) {            // wave-uniform branch
                    const half_t* gp = src0 + (size_t)ys * WWID + xst;
                    #pragma unroll
                    for (int k = 0; k < 32; ++k)
                        t[k >> 3][k & 7] = gp[(size_t)k * HWXY];
                }
                half_t* dp = &lB[(r * LBROW + xst + 1) * 32];
                #pragma unroll
                for (int v = 0; v < 4; ++v)
                    *(v8h*)(dp + v * 8) = t[v];
            }
        }
        __syncthreads();

        // ---- 9 taps, fully unrolled: shifted LDS views, immediate offsets
        #pragma unroll
        for (int tap = 0; tap < 9; ++tap) {
            const int boff = ((tap / 3) * LBROW + (tap % 3)) * 32;  // compile-time
            v16h fb0, fb1;
            {
                v8h lo = *(const v8h*)(pB0 + boff);
                v8h hi = *(const v8h*)(pB0 + boff + 8);
                #pragma unroll
                for (int i = 0; i < 8; ++i) { fb0[i] = lo[i]; fb0[i + 8] = hi[i]; }
            }
            {
                v8h lo = *(const v8h*)(pB1 + boff);
                v8h hi = *(const v8h*)(pB1 + boff + 8);
                #pragma unroll
                for (int i = 0; i < 8; ++i) { fb1[i] = lo[i]; fb1[i + 8] = hi[i]; }
            }
            #pragma unroll
            for (int mt = 0; mt < 4; ++mt) {
                const half_t* ap = pA + tap * 2048 + mt * 512;
                v8h lo = *(const v8h*)ap;
                v8h hi = *(const v8h*)(ap + 8);
                v16h fa;
                #pragma unroll
                for (int i = 0; i < 8; ++i) { fa[i] = lo[i]; fa[i + 8] = hi[i]; }
                acc[0][mt] = __builtin_amdgcn_wmma_f32_16x16x32_f16(
                    false, fa, false, fb0, (short)0, acc[0][mt], false, false);
                acc[1][mt] = __builtin_amdgcn_wmma_f32_16x16x32_f16(
                    false, fa, false, fb1, (short)0, acc[1][mt], false, false);
            }
        }
        __syncthreads();
    }

    // ---- epilogue: C layout lane->N, vgpr r -> M = r + (lane<16?0:8)
    const int mhi = (lane < 16) ? 0 : 8;
    #pragma unroll
    for (int nt = 0; nt < 2; ++nt) {
        const int x  = (wid << 5) + (nt << 4) + lrow;   // 0..255 within tile
        const int yy = y0 + (x >> 7);
        const int xi = x & 127;
        #pragma unroll
        for (int mt = 0; mt < 4; ++mt) {
            #pragma unroll
            for (int r = 0; r < 8; ++r) {
                const int cout = m0 + (mt << 4) + mhi + r;
                float v = acc[nt][mt][r];
                if (bias)  v += bias[cout];
                if (leaky) v = (v > 0.0f) ? v : 0.01f * v;
                const size_t o = ((size_t)b * Cout + cout) * HWXY + (size_t)yy * WWID + xi;
                if (out16) out16[o] = (half_t)v;
                if (out32) out32[o] = v;
            }
        }
    }
}

// ---------------------------------------------------------------------------
// Per-channel bilinear resample with absolute coords = offsets + grid
// ---------------------------------------------------------------------------
__global__ __launch_bounds__(256) void deform_sample(
    const half_t* __restrict__ hr,    // f16 [B][64][H][W]
    const half_t* __restrict__ off,   // f16 [B][128][H][W] (viewed as (B*64, HW, 2))
    half_t* __restrict__ out)         // f16 [B][64][H][W]
{
    const int idx = blockIdx.x * 256 + threadIdx.x;
    const int p = idx & (HWXY - 1);
    const int q = idx >> 14;          // b*64 + c
    const int c = q & 63;
    const int b = q >> 6;

    const size_t ob = ((size_t)b * 128 + (size_t)c * 2) * HWXY + 2 * (size_t)p;
    const float ro = (float)off[ob];
    const float co = (float)off[ob + 1];
    const int yi = p >> 7, xi = p & 127;

    float rr = fminf(fmaxf(ro + (float)yi, 0.0f), 127.0f);
    float cc = fminf(fmaxf(co + (float)xi, 0.0f), 127.0f);
    const float r0 = floorf(rr), c0 = floorf(cc);
    const float r1 = ceilf(rr),  c1f = ceilf(cc);
    const float fr = rr - r0, fc = cc - c0;
    const int r0i = (int)r0, r1i = (int)r1, c0i = (int)c0, c1i = (int)c1f;

    const half_t* img = hr + (size_t)q * HWXY;
    const float lt = (float)img[r0i * WWID + c0i];
    const float rt = (float)img[r0i * WWID + c1i];
    const float lb = (float)img[r1i * WWID + c0i];
    const float rb = (float)img[r1i * WWID + c1i];
    const float top = lt + (rt - lt) * fc;
    const float bot = lb + (rb - lb) * fc;
    out[idx] = (half_t)(top + (bot - top) * fr);
}

// ---------------------------------------------------------------------------
// Host orchestration
// ---------------------------------------------------------------------------
extern "C" void kernel_launch(void* const* d_in, const int* in_sizes, int n_in,
                              void* d_out, int out_size, void* d_ws, size_t ws_size,
                              hipStream_t stream) {
    (void)in_sizes; (void)n_in; (void)out_size; (void)ws_size;

    const size_t F = (size_t)BB * 64 * HWXY;   // elements per 64-channel map

    char* wsp = (char*)d_ws;
    auto take = [&](size_t elems) -> half_t* {
        half_t* p = (half_t*)wsp;
        wsp += elems * sizeof(half_t);
        return p;
    };
    half_t* lr16 = take(F);
    half_t* hrA  = take(F);
    half_t* hrB  = take(F);
    half_t* R1   = take(2 * F);   // e1|e2, later reused as offsets
    half_t* uA   = take(2 * F);   // est, later reused as sampled

    // per-block weights: ew1, ew2, ew3, ow, cw
    const int wCin [5] = {128, 64, 64, 128, 64};
    const int wCout[5] = { 64, 64, 128, 128, 64};
    const int widx [5] = {0, 2, 4, 6, 7};     // param slot within a block's 8 params
    half_t* w16[3][5];
    for (int blk = 0; blk < 3; ++blk)
        for (int j = 0; j < 5; ++j)
            w16[blk][j] = take((size_t)wCout[j] * wCin[j] * 9);

    auto cast = [&](const void* s, half_t* dptr, int n) {
        cast_f32_to_f16<<<(n + 255) / 256, 256, 0, stream>>>((const float*)s, dptr, n);
    };

    cast(d_in[0], lr16, (int)F);
    cast(d_in[1], hrA,  (int)F);
    for (int blk = 0; blk < 3; ++blk)
        for (int j = 0; j < 5; ++j) {
            const int n = wCout[j] * wCin[j] * 9;
            cast_permute_w<<<(n + 255) / 256, 256, 0, stream>>>(
                (const float*)d_in[2 + blk * 8 + widx[j]], w16[blk][j], wCin[j], n);
        }

    auto conv = [&](const half_t* Wt, const half_t* i0, const half_t* i1,
                    const float* bias, half_t* o16, float* o32,
                    int Cin, int split, int Cout, int leaky) {
        const int blocks = BB * 64 * (Cout >> 6);
        conv3x3_wmma<<<blocks, 256, 0, stream>>>(Wt, i0, i1, bias, o16, o32,
                                                 Cin, split, Cout, leaky);
    };

    half_t* e1   = R1;
    half_t* e2   = R1 + F;
    half_t* est  = uA;
    half_t* offb = R1;       // reuses e1|e2 after est is produced
    half_t* samp = uA;       // reuses est after offsets are produced

    const half_t* hin = hrA;
    for (int blk = 0; blk < 3; ++blk) {
        const float* eb1 = (const float*)d_in[2 + blk * 8 + 1];
        const float* eb2 = (const float*)d_in[2 + blk * 8 + 3];
        const float* eb3 = (const float*)d_in[2 + blk * 8 + 5];

        // offset estimator stack (concat handled zero-copy via split)
        conv(w16[blk][0], lr16, hin, eb1, e1,  nullptr, 128, 64, 64, 1);
        conv(w16[blk][1], e1,   e1,  eb2, e2,  nullptr,  64, 64, 64, 1);
        conv(w16[blk][2], e2,   e2,  eb3, est, nullptr,  64, 64, 128, 0);
        // ConvOffset2D offset conv
        conv(w16[blk][3], est,  est, nullptr, offb, nullptr, 128, 128, 128, 0);
        // per-channel bilinear resample of hr
        deform_sample<<<(int)(F / 256), 256, 0, stream>>>(hin, offb, samp);
        // final conv1 (no bias)
        half_t* hout = (blk == 0) ? hrB : (blk == 1) ? hrA : nullptr;
        float*  o32  = (blk == 2) ? (float*)d_out : nullptr;
        conv(w16[blk][4], samp, samp, nullptr, hout, o32, 64, 64, 64, 0);

        hin = (blk == 0) ? hrB : hrA;
    }
}
